// GNNLayer_53300544143484
// MI455X (gfx1250) — matrix operand order repaired and checked by reading, not compiled, not run on hardware
//
#include <hip/hip_runtime.h>
#include <hip/hip_bf16.h>

typedef __attribute__((ext_vector_type(16))) __bf16 v16bf;
typedef __attribute__((ext_vector_type(8)))  __bf16 v8bf;
typedef __attribute__((ext_vector_type(8)))  float  v8f;
typedef __attribute__((ext_vector_type(4)))  float  v4f;

#define WMMA_BF16(a, b, c) \
  __builtin_amdgcn_wmma_f32_16x16x32_bf16(false, (a), false, (b), (short)0, (c), false, false)

// ---------------------------------------------------------------------------
// CDNA5 async global->LDS staging (bypasses VGPRs, tracked by ASYNCcnt).
// LDS offset = low 32 bits of the flat shared-memory address.
// ---------------------------------------------------------------------------
__device__ __forceinline__ unsigned lds_off_u32(const void* p) {
  return (unsigned)(unsigned long long)p;
}

// copy 64 contiguous bytes: gbase + goff -> LDS[lds_off], via 4x b128 async
__device__ __forceinline__ void async_copy64(unsigned lds_off, const void* gbase,
                                             unsigned goff) {
  asm volatile(
      "global_load_async_to_lds_b128 %0, %1, %2\n\t"
      "global_load_async_to_lds_b128 %0, %1, %2 offset:16\n\t"
      "global_load_async_to_lds_b128 %0, %1, %2 offset:32\n\t"
      "global_load_async_to_lds_b128 %0, %1, %2 offset:48"
      :
      : "v"(lds_off), "v"(goff), "s"(gbase)
      : "memory");
}

__device__ __forceinline__ void wait_async0() {
  asm volatile("s_wait_asynccnt 0x0" ::: "memory");
}

// ---------------------------------------------------------------------------
// Fragment loaders (wave32, ISA 7.12.2 layouts)
// A (16x32, 16-bit): lanes 0-15 -> M=lane, halves [0..7]=K0..7, [8..15]=K16..23
//                    lanes16-31 -> M=lane-16, halves [0..7]=K8..15, [8..15]=K24..31
// ---------------------------------------------------------------------------
__device__ __forceinline__ v16bf load_a_f32(const float* __restrict__ row0,
                                            int ldr, int lane, int k0) {
  const int m = lane & 15, hf = lane >> 4;
  const float* r = row0 + (size_t)m * ldr + k0 + hf * 8;
  v4f a0 = *(const v4f*)(r);
  v4f a1 = *(const v4f*)(r + 4);
  v4f a2 = *(const v4f*)(r + 16);
  v4f a3 = *(const v4f*)(r + 20);
  v16bf a;
#pragma unroll
  for (int i = 0; i < 4; ++i) {
    a[i]      = (__bf16)a0[i];
    a[4 + i]  = (__bf16)a1[i];
    a[8 + i]  = (__bf16)a2[i];
    a[12 + i] = (__bf16)a3[i];
  }
  return a;
}

__device__ __forceinline__ v16bf load_a_bf16(const __bf16* __restrict__ row0,
                                             int ldr, int lane, int k0) {
  const int m = lane & 15, hf = lane >> 4;
  const __bf16* r = row0 + (size_t)m * ldr + k0 + hf * 8;
  v8bf lo = *(const v8bf*)(r);
  v8bf hi = *(const v8bf*)(r + 16);
  v16bf a;
#pragma unroll
  for (int i = 0; i < 8; ++i) { a[i] = lo[i]; a[8 + i] = hi[i]; }
  return a;
}

// B (32x16, 16-bit): lane = N (mod 16); lanes 0-15 hold K=0..7,16..23,
// lanes 16-31 hold K=8..15,24..31. stage layout: [n][k_local], 32 bf16 per col.
__device__ __forceinline__ v16bf load_b_stage(const __bf16* stage, int n0, int lane) {
  const int n = lane & 15, hf = lane >> 4;
  const __bf16* p = stage + (size_t)(n0 + n) * 32 + hf * 8;
  v8bf lo = *(const v8bf*)(p);
  v8bf hi = *(const v8bf*)(p + 16);
  v16bf b;
#pragma unroll
  for (int i = 0; i < 8; ++i) { b[i] = lo[i]; b[8 + i] = hi[i]; }
  return b;
}

// ---------------------------------------------------------------------------
// Utility kernels
// ---------------------------------------------------------------------------
__global__ void k_zero(float* p, long long n) {
  long long i = (long long)blockIdx.x * blockDim.x + threadIdx.x;
  long long st = (long long)gridDim.x * blockDim.x;
  for (; i < n; i += st) p[i] = 0.0f;
}

// w: K x Nc row-major fp32  ->  wt[n*K + k] bf16 (column-major / B-friendly)
__global__ void k_wt(const float* __restrict__ w, __bf16* __restrict__ wt,
                     int K, int Nc) {
  int idx = blockIdx.x * blockDim.x + threadIdx.x;
  if (idx >= K * Nc) return;
  int n = idx % Nc, k = idx / Nc;
  wt[(size_t)n * K + k] = (__bf16)w[idx];
}

__global__ void k_counts(const int* __restrict__ eidx, const float* __restrict__ emask,
                         float* __restrict__ counts, int N, int E, long long BE) {
  long long i = (long long)blockIdx.x * blockDim.x + threadIdx.x;
  if (i >= BE) return;
  long long b = i / E;
  int tgt = eidx[i * 2 + 1];
  atomicAdd(counts + b * N + tgt, emask[i]);
}

// ---------------------------------------------------------------------------
// Edge kernel: rel = edge_rel @ rel_w (WMMA bf16), msg = (node[src]+rel+rel_b)*mask,
// scatter atomicAdd into agg[tgt].  Block = 64 edges (4 rowG x 2 colG waves).
// ---------------------------------------------------------------------------
__global__ __launch_bounds__(256)
void k_edge(const float* __restrict__ edge_rel, const __bf16* __restrict__ wt_rel,
            const float* __restrict__ rel_b, const float* __restrict__ node,
            const int* __restrict__ eidx, const float* __restrict__ emask,
            float* __restrict__ agg, int N, int E) {
  __shared__ __bf16 stage[256 * 32];  // 16 KB K-slice of rel_w
  const int tid = threadIdx.x, lane = tid & 31, wid = tid >> 5;
  const int rg = wid >> 1, cg = wid & 1;
  const long long et = (long long)blockIdx.x * 64;
  const int b = (int)(et / E);
  const int e0 = (int)(et % E) + rg * 16;
  const float* arow = edge_rel + ((size_t)b * E + e0) * 256;
  const unsigned stage_lds = lds_off_u32(stage) + (unsigned)tid * 64u;

  v8f acc[8];
#pragma unroll
  for (int t = 0; t < 8; ++t)
#pragma unroll
    for (int j = 0; j < 8; ++j) acc[t][j] = 0.0f;

#pragma unroll
  for (int ks = 0; ks < 8; ++ks) {
    const int k0 = ks * 32;
    __syncthreads();
    // stage this thread's column (32 bf16 = 64B) asynchronously into LDS
    async_copy64(stage_lds, wt_rel, (unsigned)(tid * 256 + k0) * 2u);
    wait_async0();
    __syncthreads();
    v16bf a = load_a_f32(arow, 256, lane, k0);
#pragma unroll
    for (int t = 0; t < 8; ++t) {
      v16bf bfr = load_b_stage(stage, cg * 128 + t * 16, lane);
      acc[t] = WMMA_BF16(a, bfr, acc[t]);
    }
  }

  // epilogue: gather + mask + scatter
  const int hf = lane >> 4, nn = lane & 15;
  int srcs[8], tgts[8];
  float msks[8];
#pragma unroll
  for (int r = 0; r < 8; ++r) {
    int e = e0 + r + hf * 8;
    const int* ep = eidx + ((size_t)b * E + e) * 2;
    srcs[r] = ep[0];
    tgts[r] = ep[1];
    msks[r] = emask[(size_t)b * E + e];
  }
#pragma unroll
  for (int t = 0; t < 8; ++t) {
    const int col = cg * 128 + t * 16 + nn;
    const float rb = rel_b[col];
#pragma unroll
    for (int r = 0; r < 8; ++r) {
      float v = acc[t][r] + rb + node[((size_t)b * N + srcs[r]) * 256 + col];
      v *= msks[r];
      atomicAdd(agg + ((size_t)b * N + tgts[r]) * 256 + col, v);
    }
  }
}

// ---------------------------------------------------------------------------
// x = LayerNorm(node + agg / max(counts,1)); one wave per 256-wide row.
// ---------------------------------------------------------------------------
__global__ __launch_bounds__(256)
void k_aggln(const float* __restrict__ node, const float* __restrict__ agg,
             const float* __restrict__ counts, const float* __restrict__ ns,
             const float* __restrict__ nb, float* __restrict__ xout,
             long long rows) {
  const int lane = threadIdx.x & 31, wid = threadIdx.x >> 5;
  long long row = (long long)blockIdx.x * 8 + wid;
  if (row >= rows) return;
  const float inv = 1.0f / fmaxf(counts[row], 1.0f);
  const float* np = node + row * 256;
  const float* ap = agg + row * 256;
  float v[8], s = 0.0f, sq = 0.0f;
#pragma unroll
  for (int j = 0; j < 8; ++j) {
    float x = np[lane * 8 + j] + ap[lane * 8 + j] * inv;
    v[j] = x; s += x; sq += x * x;
  }
#pragma unroll
  for (int off = 16; off >= 1; off >>= 1) {
    s += __shfl_xor(s, off);
    sq += __shfl_xor(sq, off);
  }
  const float mean = s * (1.0f / 256.0f);
  const float rstd = rsqrtf(sq * (1.0f / 256.0f) - mean * mean + 1e-5f);
#pragma unroll
  for (int j = 0; j < 8; ++j) {
    int c = lane * 8 + j;
    xout[row * 256 + c] = (v[j] - mean) * rstd * ns[c] + nb[c];
  }
}

// ---------------------------------------------------------------------------
// Fused FFN: h = gelu(x@w1+b1) kept in 64KB LDS tile; y = h@w2+b2 + x; LN -> out.
// Block = 32 rows, 256 threads (8 waves = 2 rowG x 4 colG), 128 KB dynamic LDS.
// ---------------------------------------------------------------------------
__global__ __launch_bounds__(256)
void k_ffn(const float* __restrict__ xbuf, const __bf16* __restrict__ wt1,
           const float* __restrict__ b1, const __bf16* __restrict__ wt2,
           const float* __restrict__ b2, const float* __restrict__ fs,
           const float* __restrict__ fb, float* __restrict__ out) {
  extern __shared__ char smem[];
  __bf16* stage = (__bf16*)smem;             // 64 KB (phase1 w1 slice; phase2 w2 slice in [0,16K))
  __bf16* hsm   = (__bf16*)(smem + 65536);   // 32 x 1024 bf16 = 64 KB
  float*  ysm   = (float*)(smem + 16384);    // phase2 only: 32 x 256 f32 = 32 KB at [16K,48K)

  const int tid = threadIdx.x, lane = tid & 31, wid = tid >> 5;
  const int rg = wid >> 2, cg = wid & 3;
  const long long r0 = (long long)blockIdx.x * 32;
  const int hf = lane >> 4, nn = lane & 15;
  const unsigned stage_base = lds_off_u32(stage);

  // ---------------- phase 1: h = gelu(x @ w1 + b1) ----------------
  const float* arow = xbuf + (r0 + rg * 16) * 256;
  v8f acc[16];
#pragma unroll
  for (int t = 0; t < 16; ++t)
#pragma unroll
    for (int j = 0; j < 8; ++j) acc[t][j] = 0.0f;

#pragma unroll
  for (int ks = 0; ks < 8; ++ks) {
    const int k0 = ks * 32;
    __syncthreads();
#pragma unroll
    for (int j = 0; j < 4; ++j) {  // stage 32x1024 slice: thread copies 4 cols async
      int n = tid + j * 256;
      async_copy64(stage_base + (unsigned)n * 64u, wt1,
                   (unsigned)(n * 256 + k0) * 2u);
    }
    wait_async0();
    __syncthreads();
    v16bf a = load_a_f32(arow, 256, lane, k0);
#pragma unroll
    for (int t = 0; t < 16; ++t) {
      v16bf bfr = load_b_stage(stage, cg * 256 + t * 16, lane);
      acc[t] = WMMA_BF16(a, bfr, acc[t]);
    }
  }
#pragma unroll
  for (int t = 0; t < 16; ++t) {
    const int col = cg * 256 + t * 16 + nn;
    const float bb = b1[col];
#pragma unroll
    for (int r = 0; r < 8; ++r) {
      const int mrow = rg * 16 + r + hf * 8;
      float h = acc[t][r] + bb;
      h = 0.5f * h * (1.0f + erff(h * 0.70710678118f));  // exact gelu
      hsm[(size_t)mrow * 1024 + col] = (__bf16)h;
    }
  }

  // ---------------- phase 2: y = h @ w2 + b2 ----------------
  v8f acc2[4];
#pragma unroll
  for (int t = 0; t < 4; ++t)
#pragma unroll
    for (int j = 0; j < 8; ++j) acc2[t][j] = 0.0f;

  const __bf16* hrow = hsm + (size_t)(rg * 16) * 1024;
  for (int ks = 0; ks < 32; ++ks) {
    const int k0 = ks * 32;
    __syncthreads();
    // stage 32x256 slice of w2 (16 KB): thread copies one column async
    async_copy64(stage_base + (unsigned)tid * 64u, wt2,
                 (unsigned)(tid * 1024 + k0) * 2u);
    wait_async0();
    __syncthreads();
    v16bf a = load_a_bf16(hrow, 1024, lane, k0);
#pragma unroll
    for (int t = 0; t < 4; ++t) {
      v16bf bfr = load_b_stage(stage, cg * 64 + t * 16, lane);
      acc2[t] = WMMA_BF16(a, bfr, acc2[t]);
    }
  }
  // residual add -> ysm (disjoint from live stage[0,16K))
#pragma unroll
  for (int t = 0; t < 4; ++t) {
    const int col = cg * 64 + t * 16 + nn;
    const float bb = b2[col];
#pragma unroll
    for (int r = 0; r < 8; ++r) {
      const int mrow = rg * 16 + r + hf * 8;
      ysm[mrow * 256 + col] = acc2[t][r] + bb + xbuf[(r0 + mrow) * 256 + col];
    }
  }
  __syncthreads();

  // ---------------- final LayerNorm: each wave handles 4 rows ----------------
#pragma unroll
  for (int rr = 0; rr < 4; ++rr) {
    const int mrow = wid * 4 + rr;
    const float* yp = ysm + mrow * 256;
    float vv[8], s = 0.0f, sq = 0.0f;
#pragma unroll
    for (int j = 0; j < 8; ++j) {
      float x = yp[lane * 8 + j];
      vv[j] = x; s += x; sq += x * x;
    }
#pragma unroll
    for (int off = 16; off >= 1; off >>= 1) {
      s += __shfl_xor(s, off);
      sq += __shfl_xor(sq, off);
    }
    const float mean = s * (1.0f / 256.0f);
    const float rstd = rsqrtf(sq * (1.0f / 256.0f) - mean * mean + 1e-5f);
#pragma unroll
    for (int j = 0; j < 8; ++j) {
      int c = lane * 8 + j;
      out[(r0 + mrow) * 256 + c] = (vv[j] - mean) * rstd * fs[c] + fb[c];
    }
  }
}

// ---------------------------------------------------------------------------
extern "C" void kernel_launch(void* const* d_in, const int* in_sizes, int n_in,
                              void* d_out, int out_size, void* d_ws, size_t ws_size,
                              hipStream_t stream) {
  const float* node   = (const float*)d_in[0];
  const float* erel   = (const float*)d_in[1];
  const float* emask  = (const float*)d_in[2];
  const float* rel_w  = (const float*)d_in[3];
  const float* rel_b  = (const float*)d_in[4];
  const float* norm_s = (const float*)d_in[5];
  const float* norm_b = (const float*)d_in[6];
  const float* w1     = (const float*)d_in[7];
  const float* b1     = (const float*)d_in[8];
  const float* w2     = (const float*)d_in[9];
  const float* b2     = (const float*)d_in[10];
  const float* ffn_s  = (const float*)d_in[11];
  const float* ffn_b  = (const float*)d_in[12];
  const int*   eidx   = (const int*)d_in[13];
  float* out = (float*)d_out;

  const long long rows = (long long)in_sizes[0] / 256;  // B*N = 100000
  const long long BE   = (long long)in_sizes[13] / 2;   // B*E = 524288
  const int Bb = 2;
  const int N = (int)(rows / Bb);
  const int E = (int)(BE / Bb);

  char* ws = (char*)d_ws;
  float* agg    = (float*)ws;            // rows*256 f32
  float* counts = agg + rows * 256;      // rows f32
  size_t off = ((size_t)(rows * 257) * 4 + 255) & ~(size_t)255;
  __bf16* wt_rel = (__bf16*)(ws + off);  off += (size_t)256 * 256 * 2;
  __bf16* wt1    = (__bf16*)(ws + off);  off += (size_t)1024 * 256 * 2;
  __bf16* wt2    = (__bf16*)(ws + off);

  k_zero<<<2048, 256, 0, stream>>>(agg, rows * 257);
  k_wt<<<(256 * 256 + 255) / 256, 256, 0, stream>>>(rel_w, wt_rel, 256, 256);
  k_wt<<<(256 * 1024 + 255) / 256, 256, 0, stream>>>(w1, wt1, 256, 1024);
  k_wt<<<(1024 * 256 + 255) / 256, 256, 0, stream>>>(w2, wt2, 1024, 256);

  k_edge<<<(unsigned)(BE / 64), 256, 0, stream>>>(erel, wt_rel, rel_b, node,
                                                  eidx, emask, agg, N, E);
  k_counts<<<(unsigned)((BE + 255) / 256), 256, 0, stream>>>(eidx, emask, counts, N, E, BE);
  k_aggln<<<(unsigned)((rows + 7) / 8), 256, 0, stream>>>(node, agg, counts,
                                                          norm_s, norm_b, out, rows);
  (void)hipFuncSetAttribute(reinterpret_cast<const void*>(k_ffn),
                            hipFuncAttributeMaxDynamicSharedMemorySize, 131072);
  k_ffn<<<(unsigned)(rows / 32), 256, 131072, stream>>>(out, wt1, b1, wt2, b2,
                                                        ffn_s, ffn_b, out);
}
